// Attention_46067819217077
// MI455X (gfx1250) — matrix operand order repaired and checked by reading, not compiled, not run on hardware
//
#include <hip/hip_runtime.h>
#include <hip/hip_bf16.h>

#define EPS 1e-5f

typedef __attribute__((ext_vector_type(16))) _Float16 v16h;
typedef __attribute__((ext_vector_type(8)))  float    v8f;
typedef __attribute__((ext_vector_type(4)))  float    v4f;

// ---------------------------------------------------------------------------
// Batched NT GEMM via WMMA: C[M,N] = A[M,K] * B[N,K]^T (+ bias[N] if non-null)
// blockIdx.y in {0,1,2} selects the (B, C) pair -> fuses the Q/K/V projections
// into one launch. A, B are f32 in global memory, converted to f16 in regs.
// One wave computes a 32x64 output tile:
//   2 A fragments x 4 B fragments -> 8 x v_wmma_f32_16x16x32_f16 per K-step,
//   each B fragment reused across both M-subtiles.
// Requires: M % 32 == 0, N % 64 == 0, K % 32 == 0, blockDim.x == 256.
// ---------------------------------------------------------------------------
__global__ __launch_bounds__(256)
void wmma_gemm_nt(const float* __restrict__ A,
                  const float* __restrict__ B0, const float* __restrict__ B1,
                  const float* __restrict__ B2,
                  float* __restrict__ C0, float* __restrict__ C1,
                  float* __restrict__ C2,
                  const float* __restrict__ bias,
                  int M, int N, int K) {
    const float* __restrict__ B = (blockIdx.y == 0) ? B0 : (blockIdx.y == 1) ? B1 : B2;
    float* __restrict__       C = (blockIdx.y == 0) ? C0 : (blockIdx.y == 1) ? C1 : C2;

    const int lane = threadIdx.x & 31;
    const int wave = threadIdx.x >> 5;
    const int gw   = blockIdx.x * 8 + wave;

    const int tilesN = N >> 6;                 // 64-wide N tiles
    const int tm = gw / tilesN;                // 32-row M tile
    const int tn = gw % tilesN;
    if (tm >= (M >> 5)) return;                // wave-uniform guard (EXEC stays all-1s)

    const int r16 = lane & 15;                 // row (A/C) or column (B/C) within tile
    const int sel = lane >> 4;                 // K-half selector

    const float* __restrict__ Ap0 = A + (size_t)(tm * 32 + r16)      * (size_t)K;
    const float* __restrict__ Ap1 = A + (size_t)(tm * 32 + 16 + r16) * (size_t)K;

    v8f acc[2][4];
#pragma unroll
    for (int m = 0; m < 2; ++m)
#pragma unroll
        for (int c = 0; c < 4; ++c) acc[m][c] = (v8f){};

    for (int k0 = 0; k0 < K; k0 += 32) {
        // ---- A fragments: 16x32 f16; lane holds row r16, K = (e%8)+8*sel+16*(e/8)
        v16h af[2];
#pragma unroll
        for (int m = 0; m < 2; ++m) {
            const float* Ap = (m == 0) ? Ap0 : Ap1;
            const v4f* pa0 = (const v4f*)(Ap + k0 + 8 * sel);
            const v4f* pa1 = (const v4f*)(Ap + k0 + 16 + 8 * sel);
            v4f a0 = pa0[0], a1 = pa0[1];
            v4f a2 = pa1[0], a3 = pa1[1];
#pragma unroll
            for (int i = 0; i < 4; ++i) {
                af[m][i]      = (_Float16)a0[i];
                af[m][4 + i]  = (_Float16)a1[i];
                af[m][8 + i]  = (_Float16)a2[i];
                af[m][12 + i] = (_Float16)a3[i];
            }
        }
        __builtin_prefetch(Ap0 + k0 + 64, 0, 1);   // global_prefetch_b8

        // ---- B fragments: 32x16 f16; lane holds column r16, K = (e%8)+8*sel+16*(e/8)
#pragma unroll
        for (int c = 0; c < 4; ++c) {
            const float* __restrict__ Bp =
                B + (size_t)(tn * 64 + c * 16 + r16) * (size_t)K + k0 + 8 * sel;
            const v4f* pb = (const v4f*)Bp;
            v4f b0 = pb[0], b1 = pb[1], b2 = pb[4], b3 = pb[5];
            v16h bf;
#pragma unroll
            for (int i = 0; i < 4; ++i) {
                bf[i]      = (_Float16)b0[i];
                bf[4 + i]  = (_Float16)b1[i];
                bf[8 + i]  = (_Float16)b2[i];
                bf[12 + i] = (_Float16)b3[i];
            }
#pragma unroll
            for (int m = 0; m < 2; ++m)
                acc[m][c] = __builtin_amdgcn_wmma_f32_16x16x32_f16(
                    false, af[m], false, bf, (short)0, acc[m][c], false, false);
        }
    }

    // ---- Store: D layout — VGPR j holds row (j + 8*sel), col = r16
#pragma unroll
    for (int m = 0; m < 2; ++m) {
#pragma unroll
        for (int c = 0; c < 4; ++c) {
            const int col = tn * 64 + c * 16 + r16;
            const float bv = bias ? bias[col] : 0.0f;
#pragma unroll
            for (int j = 0; j < 8; ++j) {
                const int row = tm * 32 + m * 16 + j + 8 * sel;
                C[(size_t)row * (size_t)N + col] = acc[m][c][j] + bv;
            }
        }
    }
}

// ---------------------------------------------------------------------------
// Per-channel (row index % 2) mean/var over arrays of (2*rowsPerCh) x cols.
// blockIdx.x = tensor*2 + channel; tensor selects Y0/Y1/Y2.
// stats layout per tensor: [mean0, var0, mean1, var1] at stats + tensor*4.
// ---------------------------------------------------------------------------
__global__ __launch_bounds__(256)
void bn_stats3(const float* __restrict__ Y0, const float* __restrict__ Y1,
               const float* __restrict__ Y2,
               int rowsPerCh, int cols, float* __restrict__ stats) {
    const int tensor = blockIdx.x >> 1;
    const int ch     = blockIdx.x & 1;
    const float* __restrict__ Y = (tensor == 0) ? Y0 : (tensor == 1) ? Y1 : Y2;
    const int tid = threadIdx.x;

    float s = 0.f, ss = 0.f;
    for (int r = 0; r < rowsPerCh; ++r) {
        const float* row = Y + (size_t)(2 * r + ch) * (size_t)cols;
        for (int c = tid; c < cols; c += 256) {
            float v = row[c];
            s  += v;
            ss += v * v;
        }
    }
    __shared__ float sh0[256], sh1[256];
    sh0[tid] = s; sh1[tid] = ss;
    __syncthreads();
    for (int off = 128; off > 0; off >>= 1) {
        if (tid < off) { sh0[tid] += sh0[tid + off]; sh1[tid] += sh1[tid + off]; }
        __syncthreads();
    }
    if (tid == 0) {
        float cnt  = (float)rowsPerCh * (float)cols;
        float mean = sh0[0] / cnt;
        float var  = sh1[0] / cnt - mean * mean;
        stats[tensor * 4 + ch * 2 + 0] = mean;
        stats[tensor * 4 + ch * 2 + 1] = var;
    }
}

// ---------------------------------------------------------------------------
// Linear (kernelized) attention for n=2, per (b, h) block.
// qhat = (Y - mean)*g*rsqrt(var+eps) + beta applied on the fly via scale/shift.
// out[n,:] = (S[n,0]*v0 + S[n,1]*v1) / (S[n,0]+S[n,1])
// grid = 128 (b*h), block = 256.
// ---------------------------------------------------------------------------
__global__ __launch_bounds__(256)
void linattn(const float* __restrict__ Yq, const float* __restrict__ Yk,
             const float* __restrict__ Yv, const float* __restrict__ st,
             const float* __restrict__ gq, const float* __restrict__ bq,
             const float* __restrict__ gk, const float* __restrict__ bk,
             const float* __restrict__ gv, const float* __restrict__ bv,
             float* __restrict__ O) {
    const int b   = blockIdx.x >> 1;
    const int h   = blockIdx.x & 1;
    const int tid = threadIdx.x;

    const float* stQ = st;     // [mean0,var0,mean1,var1]
    const float* stK = st + 4;
    const float* stV = st + 8;

    const float qs0 = gq[0] * rsqrtf(stQ[1] + EPS), qo0 = bq[0] - stQ[0] * qs0;
    const float qs1 = gq[1] * rsqrtf(stQ[3] + EPS), qo1 = bq[1] - stQ[2] * qs1;
    const float ks0 = gk[0] * rsqrtf(stK[1] + EPS), ko0 = bk[0] - stK[0] * ks0;
    const float ks1 = gk[1] * rsqrtf(stK[3] + EPS), ko1 = bk[1] - stK[2] * ks1;
    const float vs0 = gv[0] * rsqrtf(stV[1] + EPS), vo0 = bv[0] - stV[0] * vs0;
    const float vs1 = gv[1] * rsqrtf(stV[3] + EPS), vo1 = bv[1] - stV[2] * vs1;

    const size_t r0 = (size_t)(b * 2 + 0) * 2048 + h * 1024;
    const size_t r1 = (size_t)(b * 2 + 1) * 2048 + h * 1024;
    const float* q0p = Yq + r0; const float* q1p = Yq + r1;
    const float* k0p = Yk + r0; const float* k1p = Yk + r1;
    const float* v0p = Yv + r0; const float* v1p = Yv + r1;

    float s00 = 0.f, s01 = 0.f, s10 = 0.f, s11 = 0.f;
    for (int d = tid; d < 1024; d += 256) {
        float q0 = q0p[d] * qs0 + qo0, q1 = q1p[d] * qs1 + qo1;
        float k0 = k0p[d] * ks0 + ko0, k1 = k1p[d] * ks1 + ko1;
        s00 += q0 * k0; s01 += q0 * k1;
        s10 += q1 * k0; s11 += q1 * k1;
    }

    __shared__ float sh[4][256];
    sh[0][tid] = s00; sh[1][tid] = s01; sh[2][tid] = s10; sh[3][tid] = s11;
    __syncthreads();
    for (int off = 128; off > 0; off >>= 1) {
        if (tid < off) {
#pragma unroll
            for (int i = 0; i < 4; ++i) sh[i][tid] += sh[i][tid + off];
        }
        __syncthreads();
    }
    const float S00 = sh[0][0], S01 = sh[1][0], S10 = sh[2][0], S11 = sh[3][0];
    const float di0 = 1.0f / (S00 + S01);
    const float di1 = 1.0f / (S10 + S11);
    const float w00 = S00 * di0, w01 = S01 * di0;
    const float w10 = S10 * di1, w11 = S11 * di1;

    float* o0 = O + r0;
    float* o1 = O + r1;
    for (int e = tid; e < 1024; e += 256) {
        float v0 = v0p[e] * vs0 + vo0;
        float v1 = v1p[e] * vs1 + vo1;
        o0[e] = w00 * v0 + w01 * v1;
        o1[e] = w10 * v0 + w11 * v1;
    }
}

// ---------------------------------------------------------------------------
// Final BatchNorm2d apply: out = (Z - mean)*g*rsqrt(var+eps) + beta, per channel.
// ---------------------------------------------------------------------------
__global__ __launch_bounds__(256)
void bn_apply(const float* __restrict__ Z, const float* __restrict__ st,
              const float* __restrict__ g, const float* __restrict__ bta,
              float* __restrict__ out, int cols, int total) {
    int idx = blockIdx.x * 256 + threadIdx.x;
    if (idx >= total) return;
    int n = (idx / cols) & 1;
    float sc = g[n] * rsqrtf(st[n * 2 + 1] + EPS);
    float sh = bta[n] - st[n * 2] * sc;
    out[idx] = Z[idx] * sc + sh;
}

// ---------------------------------------------------------------------------
extern "C" void kernel_launch(void* const* d_in, const int* in_sizes, int n_in,
                              void* d_out, int out_size, void* d_ws, size_t ws_size,
                              hipStream_t stream) {
    const float* x    = (const float*)d_in[0];   // (64,2,32,32) -> (128,1024)
    const float* Wq   = (const float*)d_in[1];   // (2048,1024)
    const float* Wk   = (const float*)d_in[2];
    const float* Wv   = (const float*)d_in[3];
    const float* Wo   = (const float*)d_in[4];   // (1024,2048)
    const float* bo   = (const float*)d_in[5];   // (1024,)
    const float* g_q  = (const float*)d_in[6];
    const float* b_q  = (const float*)d_in[7];
    const float* g_k  = (const float*)d_in[8];
    const float* b_k  = (const float*)d_in[9];
    const float* g_v  = (const float*)d_in[10];
    const float* b_v  = (const float*)d_in[11];
    const float* g_bn = (const float*)d_in[12];
    const float* b_bn = (const float*)d_in[13];

    float* ws   = (float*)d_ws;
    float* Yq   = ws;                    // 128*2048
    float* Yk   = Yq + 128 * 2048;       // 128*2048
    float* Yv   = Yk + 128 * 2048;       // 128*2048
    float* O    = Yv + 128 * 2048;       // 128*2048
    float* Z    = O  + 128 * 2048;       // 128*1024
    float* stQKV = Z + 128 * 1024;       // 12 (q:4, k:4, v:4)
    float* stZ   = stQKV + 12;           // 4

    // Q/K/V projections fused in one launch: M=128, N=2048, K=1024
    // 32-row x 64-col wave tiles -> (4*32) waves / 8 = 16 blocks, grid.y = 3
    wmma_gemm_nt<<<dim3(16, 3), 256, 0, stream>>>(
        x, Wq, Wk, Wv, Yq, Yk, Yv, nullptr, 128, 2048, 1024);

    // BatchNorm1d stats for q, k, v in one launch (3 tensors x 2 channels)
    bn_stats3<<<6, 256, 0, stream>>>(Yq, Yk, Yv, 64, 2048, stQKV);

    // Kernelized attention (n=2 closed form), BN folded in
    linattn<<<128, 256, 0, stream>>>(Yq, Yk, Yv, stQKV,
                                     g_q, b_q, g_k, b_k, g_v, b_v, O);

    // Output projection: M=128, N=1024, K=2048 -> (4*16) waves / 8 = 8 blocks
    wmma_gemm_nt<<<dim3(8, 1), 256, 0, stream>>>(
        O, Wo, Wo, Wo, Z, Z, Z, bo, 128, 1024, 2048);

    // Final BatchNorm2d
    bn_stats3<<<2, 256, 0, stream>>>(Z, Z, Z, 64, 1024, stZ);
    bn_apply<<<512, 256, 0, stream>>>(Z, stZ, g_bn, b_bn, (float*)d_out,
                                      1024, 128 * 1024);
}
